// NL_44641890074803
// MI455X (gfx1250) — matrix operand order repaired and checked by reading, not compiled, not run on hardware
//
#include <hip/hip_runtime.h>
#include <hip/hip_bf16.h>
#include <stdint.h>
#include <stddef.h>

#define B_   64
#define C_   256
#define L_   4096
#define OUT_ 256

typedef __attribute__((ext_vector_type(16))) __bf16 v16bf;
typedef __attribute__((ext_vector_type(8)))  float  v8f;

struct __attribute__((aligned(16))) U4 { unsigned x, y, z, w; };
union Frag { U4 q[2]; v16bf v; };

static __device__ inline unsigned short f2bf(float f) {
  union { __bf16 h; unsigned short u; } c; c.h = (__bf16)f; return c.u;
}
static __device__ inline float bf2f(unsigned short u) {
  union { unsigned short u; __bf16 h; } c; c.u = u; return (float)c.h;
}

static __device__ inline v8f wmma_bf16(const Frag& a, const Frag& b, v8f c) {
  // D = A(16x32 bf16) x B(32x16 bf16) + C(16x16 f32)
  return __builtin_amdgcn_wmma_f32_16x16x32_bf16(
      /*neg_a=*/false, a.v, /*neg_b=*/false, b.v,
      /*c_mod=*/(short)0, c, /*reuse_a=*/false, /*reuse_b=*/false);
}

// ---------------------------------------------------------------------------
// K0: convert the five weight matrices f32 -> bf16 into ws; zero stats buffer.
// ---------------------------------------------------------------------------
__global__ __launch_bounds__(256) void k0_convert(
    const float* __restrict__ wq, const float* __restrict__ wk,
    const float* __restrict__ wv, const float* __restrict__ wc,
    const float* __restrict__ wo, unsigned short* __restrict__ wbf,
    float* __restrict__ stats) {
  const int i = blockIdx.x * 256 + threadIdx.x;
  const int n = C_ * C_;
  if (i < n) {
    wbf[0 * n + i] = f2bf(wq[i]);
    wbf[1 * n + i] = f2bf(wk[i]);
    wbf[2 * n + i] = f2bf(wv[i]);
    wbf[3 * n + i] = f2bf(wc[i]);
    wbf[4 * n + i] = f2bf(wo[i]);
  }
  if (i < 4 * B_) stats[i] = 0.0f;   // sum, sumsq, mean, rstd
}

// ---------------------------------------------------------------------------
// K1: fused QKV GEMM.  Per block: fixed b, 64 l's.  x tile staged (transposed,
// bf16) in LDS once and reused for Wq, Wk, Wv.  Outputs in [l][b][c] bf16.
// ---------------------------------------------------------------------------
__global__ __launch_bounds__(256) void k1_qkv(
    const float* __restrict__ x, const unsigned short* __restrict__ wbf,
    unsigned short* __restrict__ q, unsigned short* __restrict__ k,
    unsigned short* __restrict__ v) {
  const int b  = blockIdx.y;
  const int l0 = blockIdx.x * 64;
  __shared__ unsigned short xs[64 * 264];        // [l][c], pitch 264 (bank-rotating)
  const int t = threadIdx.x, lane = t & 31, wave = t >> 5;

  // Stage x[b][c][l0..l0+63] -> xs[l][c] (coalesced f32 loads along l).
  for (int it = 0; it < 16; ++it) {
    const int c  = it * 16 + (t >> 4);
    const int ch = t & 15;
    const float4 xv = *(const float4*)(x + (((size_t)b * C_ + c) * L_ + l0 + ch * 4));
    xs[(ch * 4 + 0) * 264 + c] = f2bf(xv.x);
    xs[(ch * 4 + 1) * 264 + c] = f2bf(xv.y);
    xs[(ch * 4 + 2) * 264 + c] = f2bf(xv.z);
    xs[(ch * 4 + 3) * 264 + c] = f2bf(xv.w);
  }
  __syncthreads();

  const int mlane = lane & 15, half = lane >> 4;
  for (int g = 0; g < 3; ++g) {
    const unsigned short* w = wbf + (size_t)g * C_ * C_;
    unsigned short* dst = (g == 0) ? q : (g == 1) ? k : v;
    for (int mi = 0; mi < 2; ++mi) {
      const int mt = wave * 2 + mi;
      const int m  = mt * 16 + mlane;
      __builtin_prefetch(w + (size_t)m * C_, 0, 2);
      for (int nt = 0; nt < 4; ++nt) {
        const int n = nt * 16 + mlane;
        v8f acc = {};
        for (int kk = 0; kk < 8; ++kk) {
          Frag a, bf;
          const int kbA = kk * 32 + half * 8;           // A: lane holds K chunks kbA, kbA+16
          a.q[0] = *(const U4*)(w + (size_t)m * C_ + kbA);
          a.q[1] = *(const U4*)(w + (size_t)m * C_ + kbA + 16);
          const int kbB = kk * 32 + half * 8;           // B: lane holds K chunks kbB, kbB+16
          bf.q[0] = *(const U4*)(&xs[n * 264 + kbB]);
          bf.q[1] = *(const U4*)(&xs[n * 264 + kbB + 16]);
          acc = wmma_bf16(a, bf, acc);
        }
        // D: lane = column n (-> l), rows co = mt*16 + half*8 + r (contiguous)
        unsigned short o[8] __attribute__((aligned(16)));
        for (int r = 0; r < 8; ++r) o[r] = f2bf(acc[r]);
        const size_t l   = (size_t)(l0 + n);
        const size_t off = (l * B_ + b) * C_ + mt * 16 + half * 8;
        *(U4*)(dst + off) = *(const U4*)o;
      }
    }
  }
}

// ---------------------------------------------------------------------------
// K2: per-l batch attention.  att = softmax_d(q k^T / 16); virt = att v
// (computed as virt^T = v^T att^T so D rows land on contiguous c).
// q/k fragments come straight from global ([l][b][c] gives contiguous K runs).
// Accumulates per-b sum / sumsq for GroupNorm via f32 atomics.
// ---------------------------------------------------------------------------
__global__ __launch_bounds__(256) void k2_attn(
    const unsigned short* __restrict__ q, const unsigned short* __restrict__ k,
    const unsigned short* __restrict__ v, unsigned short* __restrict__ virt,
    float* __restrict__ sum, float* __restrict__ sumsq) {
  const int l = blockIdx.x;
  __shared__ float          att_s[64 * 68];      // f32 att, pitch 68
  __shared__ unsigned short attb [64 * 72];      // bf16 att [b][d], pitch 72
  __shared__ unsigned short vt   [256 * 72];     // v transposed [c][d], pitch 72
  const int t = threadIdx.x, lane = t & 31;
  const int wave = t >> 5, mlane = lane & 15, half = lane >> 4;
  const size_t base = (size_t)l * B_ * C_;

  // Stage v[l][d][c] -> vt[c][d].
  for (int it = 0; it < 8; ++it) {
    const int idx = it * 256 + t;
    const int d = idx >> 5, c0 = (idx & 31) * 8;
    U4 r = *(const U4*)(v + base + (size_t)d * C_ + c0);
    const unsigned short* rs = (const unsigned short*)&r;
    for (int j = 0; j < 8; ++j) vt[(c0 + j) * 72 + d] = rs[j];
  }

  // GEMM1: att[b][d] = sum_c q[b][c] * k[d][c], scaled by 1/sqrt(C)=0.0625.
  for (int ti = 0; ti < 2; ++ti) {
    const int idx = wave * 2 + ti, mt = idx >> 2, nt = idx & 3;
    const int m = mt * 16 + mlane, n = nt * 16 + mlane;
    v8f acc = {};
    for (int kk = 0; kk < 8; ++kk) {
      Frag a, bf;
      const int kbA = kk * 32 + half * 8;
      a.q[0] = *(const U4*)(q + base + (size_t)m * C_ + kbA);
      a.q[1] = *(const U4*)(q + base + (size_t)m * C_ + kbA + 16);
      const int kbB = kk * 32 + half * 8;    // B[c][d] = k[d][c]: lane n=d reads row d
      bf.q[0] = *(const U4*)(k + base + (size_t)n * C_ + kbB);
      bf.q[1] = *(const U4*)(k + base + (size_t)n * C_ + kbB + 16);
      acc = wmma_bf16(a, bf, acc);
    }
    for (int r = 0; r < 8; ++r)
      att_s[(mt * 16 + half * 8 + r) * 68 + n] = acc[r] * 0.0625f;
  }
  __syncthreads();

  // Softmax over d per row b: 4 lanes per row, shuffle reductions.
  {
    const int row = t >> 2, jj = (t & 3) * 16;
    float mx = -3.4e38f;
    for (int d = 0; d < 16; ++d) mx = fmaxf(mx, att_s[row * 68 + jj + d]);
    mx = fmaxf(mx, __shfl_xor(mx, 1, 32));
    mx = fmaxf(mx, __shfl_xor(mx, 2, 32));
    float s = 0.0f;
    for (int d = 0; d < 16; ++d) {
      const float e = __expf(att_s[row * 68 + jj + d] - mx);
      att_s[row * 68 + jj + d] = e;
      s += e;
    }
    s += __shfl_xor(s, 1, 32);
    s += __shfl_xor(s, 2, 32);
    const float inv = 1.0f / s;
    for (int d = 0; d < 16; ++d)
      attb[row * 72 + jj + d] = f2bf(att_s[row * 68 + jj + d] * inv);
  }
  __syncthreads();

  // GEMM2 (swapped): virt^T[c][b] = sum_d vt[c][d] * attb[b][d].
  // A = vt (M=c, K=d), B = att^T (K=d, N=b): both layouts already match.
  // D: lane column = b (fixed per wave), rows = contiguous c -> b128 stores.
  float psum = 0.0f, psq = 0.0f;
  const int nt = wave >> 1;                       // b-tile, fixed per wave
  const int bcol = nt * 16 + mlane;               // this lane's b column
  for (int ti = 0; ti < 8; ++ti) {
    const int mt = (wave & 1) * 8 + ti;           // c-tile
    const int m = mt * 16 + mlane;
    v8f acc = {};
    for (int kk = 0; kk < 2; ++kk) {
      Frag a, bf;
      const int kbA = kk * 32 + half * 8;
      a.q[0] = *(const U4*)(&vt[m * 72 + kbA]);
      a.q[1] = *(const U4*)(&vt[m * 72 + kbA + 16]);
      const int kbB = kk * 32 + half * 8;
      bf.q[0] = *(const U4*)(&attb[bcol * 72 + kbB]);
      bf.q[1] = *(const U4*)(&attb[bcol * 72 + kbB + 16]);
      acc = wmma_bf16(a, bf, acc);
    }
    unsigned short o[8] __attribute__((aligned(16)));
    for (int r = 0; r < 8; ++r) {
      const float vv = acc[r];
      psum += vv; psq += vv * vv;
      o[r] = f2bf(vv);
    }
    const int c0 = mt * 16 + half * 8;
    *(U4*)(virt + ((size_t)l * B_ + bcol) * C_ + c0) = *(const U4*)o;
  }
  // Lanes t and t^16 share the same b column; combine, then one atomic per b.
  psum += __shfl_xor(psum, 16, 32);
  psq  += __shfl_xor(psq,  16, 32);
  if (half == 0) {
    atomicAdd(&sum[bcol],   psum);
    atomicAdd(&sumsq[bcol], psq);
  }
}

// ---------------------------------------------------------------------------
// K3: finalize GroupNorm statistics.
// ---------------------------------------------------------------------------
__global__ void k3_stats(const float* __restrict__ sum, const float* __restrict__ sumsq,
                         float* __restrict__ mean, float* __restrict__ rstd) {
  const int b = threadIdx.x;
  if (b < B_) {
    const float inv = 1.0f / ((float)C_ * (float)L_);
    const float m   = sum[b] * inv;
    const float var = sumsq[b] * inv - m * m;
    mean[b] = m;
    rstd[b] = rsqrtf(var + 1e-5f);
  }
}

// ---------------------------------------------------------------------------
// K4: GroupNorm-affine-ReLU -> Wc GEMM (+x residual, coalesced) -> Wout GEMM.
// Both GEMMs computed transposed (M = l) so the x loads and out stores are
// contiguous float4 pairs.  y lives only in LDS (bf16).  Per block: b, 32 l's.
// ---------------------------------------------------------------------------
__global__ __launch_bounds__(256) void k4_out(
    const unsigned short* __restrict__ virt, const float* __restrict__ x,
    const unsigned short* __restrict__ wbf,
    const float* __restrict__ gamma, const float* __restrict__ beta,
    const float* __restrict__ mean, const float* __restrict__ rstd,
    float* __restrict__ out) {
  const int b  = blockIdx.y;
  const int l0 = blockIdx.x * 32;
  __shared__ unsigned short vn[32 * 264];        // normalized virt, [l][c]
  __shared__ unsigned short ys[32 * 264];        // y = x + Wc*vn, [l][co]
  const int t = threadIdx.x, lane = t & 31;
  const int wave = t >> 5, mlane = lane & 15, half = lane >> 4;
  const float mu = mean[b], rs = rstd[b];

  // Stage + normalize: vn[l][c] = relu((virt - mu)*rs*gamma + beta) in bf16.
  for (int it = 0; it < 4; ++it) {
    const int idx = it * 256 + t;
    const int li = idx >> 5, c0 = (idx & 31) * 8;
    U4 r = *(const U4*)(virt + (((size_t)(l0 + li)) * B_ + b) * C_ + c0);
    const unsigned short* rp = (const unsigned short*)&r;
    unsigned short o[8] __attribute__((aligned(16)));
    for (int j = 0; j < 8; ++j) {
      float f = bf2f(rp[j]);
      f = (f - mu) * rs * gamma[c0 + j] + beta[c0 + j];
      o[j] = f2bf(fmaxf(f, 0.0f));
    }
    *(U4*)(&vn[li * 264 + c0]) = *(const U4*)o;
  }
  __syncthreads();

  const unsigned short* wc = wbf + 3 * (size_t)C_ * C_;
  const unsigned short* wo = wbf + 4 * (size_t)C_ * C_;

  // GEMM1 (swapped): y^T[l][co] = sum_c vn[l][c] * wc[co][c]  (+ x residual).
  // A = vn (M=l, K=c); B = Wc^T (K=c, N=co) = wc rows.  D rows = contiguous l
  // -> residual x[b][co][l..l+7] is two coalesced float4 loads per lane.
  for (int ti = 0; ti < 4; ++ti) {
    const int idx = wave * 4 + ti, mt = idx >> 4, ntc = idx & 15;
    const int m = mt * 16 + mlane;                // local l row
    const int co = ntc * 16 + mlane;              // this lane's co column
    v8f acc = {};
    for (int kk = 0; kk < 8; ++kk) {
      Frag a, bf;
      const int kbA = kk * 32 + half * 8;
      a.q[0] = *(const U4*)(&vn[m * 264 + kbA]);
      a.q[1] = *(const U4*)(&vn[m * 264 + kbA + 16]);
      const int kbB = kk * 32 + half * 8;
      bf.q[0] = *(const U4*)(wc + (size_t)co * C_ + kbB);
      bf.q[1] = *(const U4*)(wc + (size_t)co * C_ + kbB + 16);
      acc = wmma_bf16(a, bf, acc);
    }
    const int lrow = mt * 16 + half * 8;          // first of 8 contiguous l's
    const float* xp = x + ((size_t)b * C_ + co) * L_ + l0 + lrow;
    const float4 x0 = *(const float4*)xp;
    const float4 x1 = *(const float4*)(xp + 4);
    const float xr[8] = {x0.x, x0.y, x0.z, x0.w, x1.x, x1.y, x1.z, x1.w};
    for (int r = 0; r < 8; ++r)
      ys[(lrow + r) * 264 + co] = f2bf(acc[r] + xr[r]);
  }
  __syncthreads();

  // GEMM2 (swapped): out^T[l][o] = sum_co ys[l][co] * wout[o][co].
  // D rows = contiguous l -> two coalesced float4 stores per lane.
  for (int ti = 0; ti < 4; ++ti) {
    const int idx = wave * 4 + ti, mt = idx >> 4, nto = idx & 15;
    const int m = mt * 16 + mlane;                // local l row
    const int oc = nto * 16 + mlane;              // this lane's o column
    v8f acc = {};
    for (int kk = 0; kk < 8; ++kk) {
      Frag a, bf;
      const int kbA = kk * 32 + half * 8;
      a.q[0] = *(const U4*)(&ys[m * 264 + kbA]);
      a.q[1] = *(const U4*)(&ys[m * 264 + kbA + 16]);
      const int kbB = kk * 32 + half * 8;
      bf.q[0] = *(const U4*)(wo + (size_t)oc * C_ + kbB);
      bf.q[1] = *(const U4*)(wo + (size_t)oc * C_ + kbB + 16);
      acc = wmma_bf16(a, bf, acc);
    }
    const int lrow = mt * 16 + half * 8;
    float* op = out + ((size_t)b * OUT_ + oc) * L_ + l0 + lrow;
    float4 o0; o0.x = acc[0]; o0.y = acc[1]; o0.z = acc[2]; o0.w = acc[3];
    float4 o1; o1.x = acc[4]; o1.y = acc[5]; o1.z = acc[6]; o1.w = acc[7];
    *(float4*)op       = o0;
    *(float4*)(op + 4) = o1;
  }
}

// ---------------------------------------------------------------------------
extern "C" void kernel_launch(void* const* d_in, const int* in_sizes, int n_in,
                              void* d_out, int out_size, void* d_ws, size_t ws_size,
                              hipStream_t stream) {
  const float* x     = (const float*)d_in[0];
  const float* Wq    = (const float*)d_in[1];
  const float* Wk    = (const float*)d_in[2];
  const float* Wv    = (const float*)d_in[3];
  const float* Wc    = (const float*)d_in[4];
  const float* Wout  = (const float*)d_in[5];
  const float* gamma = (const float*)d_in[6];
  const float* beta  = (const float*)d_in[7];
  float* out = (float*)d_out;

  char* ws = (char*)d_ws;
  unsigned short* wbf = (unsigned short*)ws;                   // 5*C*C bf16
  float* stats = (float*)(ws + (size_t)5 * C_ * C_ * 2);       // 4*B f32
  float* sum = stats, *sumsq = stats + B_, *mean = stats + 2 * B_, *rstd = stats + 3 * B_;

  const size_t qkv_elems = (size_t)L_ * B_ * C_;               // 64M bf16 each
  unsigned short* qb   = (unsigned short*)(ws + (1u << 20));
  unsigned short* kb   = qb + qkv_elems;
  unsigned short* vb   = kb + qkv_elems;
  unsigned short* virt = vb + qkv_elems;                       // total ~513 MB in ws

  k0_convert<<<(C_ * C_ + 255) / 256, 256, 0, stream>>>(Wq, Wk, Wv, Wc, Wout, wbf, stats);

  dim3 g1(L_ / 64, B_);
  k1_qkv<<<g1, 256, 0, stream>>>(x, wbf, qb, kb, vb);

  k2_attn<<<L_, 256, 0, stream>>>(qb, kb, vb, virt, sum, sumsq);

  k3_stats<<<1, 64, 0, stream>>>(sum, sumsq, mean, rstd);

  dim3 g4(L_ / 32, B_);
  k4_out<<<g4, 256, 0, stream>>>(virt, x, wbf, gamma, beta, mean, rstd, out);
}
